// ArcNegFace_10179072492041
// MI455X (gfx1250) — compile-verified, matrix operand order checked by reading, not run on hardware
//
#include <hip/hip_runtime.h>

// ---------------- problem constants (from reference) ----------------
#define IN_F   512
#define OUT_F  100000
#define BATCH  512

#define BM 128            // block tile M
#define BN 128            // block tile N
#define BK 32             // K chunk (one f16 WMMA depth)
#define LDSTR 40          // LDS row stride in halves (80B: 16B-aligned, bank-spread)

typedef _Float16 v16h __attribute__((ext_vector_type(16)));
typedef _Float16 v8h  __attribute__((ext_vector_type(8)));
typedef float    v8f  __attribute__((ext_vector_type(8)));

// arcface constants
__device__ __constant__ const float kSCALE  = 64.0f;
__device__ __constant__ const float kALPHA  = 1.2f;
__device__ __constant__ const float kINVSIG = 0.5f;                  // 1/SIGMA
__device__ __constant__ const float kCOSM   = 0.8775825618903728f;   // cos(0.5)
__device__ __constant__ const float kSINM   = 0.4794255386042030f;   // sin(0.5)
__device__ __constant__ const float kTHRESH = -0.8775825618903728f;  // cos(pi-0.5)
__device__ __constant__ const float kMM     = 0.2397127693021015f;   // sin(0.5)*0.5

// ------------------------------------------------------------------
// Pass A: per-row 1/||x||, and margined target from cos(x, w[label])
// ------------------------------------------------------------------
__global__ __launch_bounds__(128) void arcneg_prep(
    const float* __restrict__ x, const int* __restrict__ label,
    const float* __restrict__ w, float* __restrict__ ws)
{
    __shared__ float s0[128], s1[128], s2[128];
    const int b   = blockIdx.x;
    const int tid = threadIdx.x;
    const int lbl = label[b];
    const float* xr = x + (size_t)b   * IN_F;
    const float* wr = w + (size_t)lbl * IN_F;

    float sx2 = 0.f, sxw = 0.f, sw2 = 0.f;
    for (int j = tid * 4; j < IN_F; j += 128 * 4) {
        float4 xv = *(const float4*)(xr + j);
        float4 wv = *(const float4*)(wr + j);
        sx2 += xv.x*xv.x + xv.y*xv.y + xv.z*xv.z + xv.w*xv.w;
        sw2 += wv.x*wv.x + wv.y*wv.y + wv.z*wv.z + wv.w*wv.w;
        sxw += xv.x*wv.x + xv.y*wv.y + xv.z*wv.z + xv.w*wv.w;
    }
    s0[tid] = sx2; s1[tid] = sxw; s2[tid] = sw2;
    __syncthreads();
    for (int s = 64; s > 0; s >>= 1) {
        if (tid < s) { s0[tid] += s0[tid+s]; s1[tid] += s1[tid+s]; s2[tid] += s2[tid+s]; }
        __syncthreads();
    }
    if (tid == 0) {
        float xinv = rsqrtf(s0[0]);
        float winv = rsqrtf(s2[0]);
        float cl   = s1[0] * xinv * winv;
        float tgt  = (cl > kTHRESH)
                   ? (cl * kCOSM - sqrtf(fmaxf(1.f - cl*cl, 0.f)) * kSINM)
                   : (cl - kMM);
        ws[b]         = xinv;  // 1/||x_b||
        ws[BATCH + b] = tgt;   // target_b
    }
}

// LDS fragment loader: 16-bit A/B fragment layout for V_WMMA_*_16X16X32.
// lane = row (A) / col (B) in [0,16); kb = (lane>>4)*8 selects K phase.
// v16h[0..7] = K kb+0..7 ; v16h[8..15] = K kb+16..23.
__device__ __forceinline__ v16h load_frag(const _Float16* p, int row, int kb)
{
    const _Float16* base = p + row * LDSTR + kb;
    v8h lo = *(const v8h*)(base);        // K = kb .. kb+7
    v8h hi = *(const v8h*)(base + 16);   // K = kb+16 .. kb+23
    return __builtin_shufflevector(lo, hi, 0,1,2,3,4,5,6,7,8,9,10,11,12,13,14,15);
}

// ------------------------------------------------------------------
// Pass B: fused cosine GEMM (split-f16 WMMA, f32 accumulate),
//         weight norms computed during the streaming load, epilogue.
// ------------------------------------------------------------------
__global__ __launch_bounds__(256) void arcneg_gemm(
    const float* __restrict__ x, const float* __restrict__ w,
    const float* __restrict__ ws, float* __restrict__ out)
{
    __shared__ _Float16 sA_hi[BM * LDSTR] __attribute__((aligned(16)));
    __shared__ _Float16 sA_lo[BM * LDSTR] __attribute__((aligned(16)));
    __shared__ _Float16 sB_hi[BN * LDSTR] __attribute__((aligned(16)));
    __shared__ _Float16 sB_lo[BN * LDSTR] __attribute__((aligned(16)));
    __shared__ float sRed[256];
    __shared__ float sWinv[BN];
    __shared__ float sXinv[BM];
    __shared__ float sTgt[BM];

    const int tid  = threadIdx.x;
    const int lane = tid & 31;
    const int wid  = tid >> 5;          // 8 waves
    const int wm   = wid & 3;           // 4 waves along M -> 32 rows each
    const int wn   = wid >> 2;          // 2 waves along N -> 64 cols each
    const int b0   = blockIdx.x * BM;   // M block (x fastest: weight-tile L2 reuse)
    const int c0   = blockIdx.y * BN;   // N block

    if (tid < BM) {
        sXinv[tid] = ws[b0 + tid];
        sTgt[tid]  = ws[BATCH + b0 + tid];
    }

    // staging mapping: thread -> (tile row r, 16-float half h)
    const int r = tid >> 1;
    const int h = tid & 1;
    const float* aRow = x + (size_t)(b0 + r) * IN_F + h * 16;
    int cr = c0 + r; if (cr > OUT_F - 1) cr = OUT_F - 1;     // clamp ragged N edge
    const float* bRow = w + (size_t)cr * IN_F + h * 16;

    v8f acc[2][4] = {};
    float w2 = 0.f;

    for (int k0 = 0; k0 < IN_F; k0 += BK) {
        __syncthreads();
        // ---- stage 128x32 f32 tiles of X and W, split to hi/lo f16 planes ----
        float bufA[16], bufB[16];
        #pragma unroll
        for (int j = 0; j < 4; ++j) {
            *(float4*)(bufA + 4*j) = *(const float4*)(aRow + k0 + 4*j);
            *(float4*)(bufB + 4*j) = *(const float4*)(bRow + k0 + 4*j);
        }
        #pragma unroll
        for (int part = 0; part < 2; ++part) {
            v8h ah, al, bh, bl;
            #pragma unroll
            for (int j = 0; j < 8; ++j) {
                float va = bufA[part*8 + j];
                _Float16 vah = (_Float16)va;
                ah[j] = vah; al[j] = (_Float16)(va - (float)vah);
                float vb = bufB[part*8 + j];
                _Float16 vbh = (_Float16)vb;
                bh[j] = vbh; bl[j] = (_Float16)(vb - (float)vbh);
                w2 += vb * vb;                     // fused ||w_c||^2
            }
            const int off = r * LDSTR + h * 16 + part * 8;   // 16B aligned
            *(v8h*)(sA_hi + off) = ah;  *(v8h*)(sA_lo + off) = al;
            *(v8h*)(sB_hi + off) = bh;  *(v8h*)(sB_lo + off) = bl;
        }
        __syncthreads();

        // ---- WMMA: D += Ahi*Bhi + Alo*Bhi + Ahi*Blo (f32 accumulate) ----
        const int kb = (lane >> 4) * 8;
        const int rm = lane & 15;
        v16h Ah[2], Al[2];
        #pragma unroll
        for (int mt = 0; mt < 2; ++mt) {
            const int row = wm * 32 + mt * 16 + rm;
            Ah[mt] = load_frag(sA_hi, row, kb);
            Al[mt] = load_frag(sA_lo, row, kb);
        }
        #pragma unroll
        for (int nt = 0; nt < 4; ++nt) {
            const int col = wn * 64 + nt * 16 + rm;
            v16h Bh = load_frag(sB_hi, col, kb);
            v16h Bl = load_frag(sB_lo, col, kb);
            #pragma unroll
            for (int mt = 0; mt < 2; ++mt) {
                acc[mt][nt] = __builtin_amdgcn_wmma_f32_16x16x32_f16(
                    false, Ah[mt], false, Bh, (short)0, acc[mt][nt], false, false);
                acc[mt][nt] = __builtin_amdgcn_wmma_f32_16x16x32_f16(
                    false, Al[mt], false, Bh, (short)0, acc[mt][nt], false, false);
                acc[mt][nt] = __builtin_amdgcn_wmma_f32_16x16x32_f16(
                    false, Ah[mt], false, Bl, (short)0, acc[mt][nt], false, false);
            }
        }
    }

    // ---- reduce per-row ||w||^2 partials (threads 2r, 2r+1 share row r) ----
    __syncthreads();
    sRed[tid] = w2;
    __syncthreads();
    if (tid < BN) sWinv[tid] = rsqrtf(sRed[2*tid] + sRed[2*tid + 1]);
    __syncthreads();

    // ---- epilogue: cos -> gaussian-reweighted negative logits ----
    const int rm    = lane & 15;
    const int mhalf = (lane >> 4) * 8;   // C/D layout: VGPR rr -> M = rr + mhalf
    #pragma unroll
    for (int nt = 0; nt < 4; ++nt) {
        const int cb = wn * 64 + nt * 16 + rm;
        const int c  = c0 + cb;
        if (c < OUT_F) {
            const float winv = sWinv[cb];
            #pragma unroll
            for (int mt = 0; mt < 2; ++mt) {
                #pragma unroll
                for (int rr = 0; rr < 8; ++rr) {
                    const int rwl = wm * 32 + mt * 16 + mhalf + rr;
                    float cosv = acc[mt][nt][rr] * sXinv[rwl] * winv;
                    float d    = cosv - sTgt[rwl];
                    float t    = kALPHA * __expf(-d * d * kINVSIG);
                    out[(size_t)(b0 + rwl) * OUT_F + c] = kSCALE * (t * cosv + t - 1.f);
                }
            }
        }
    }
}

// ------------------------------------------------------------------
// Pass C: overwrite the single label element per row with SCALE*target
// ------------------------------------------------------------------
__global__ __launch_bounds__(512) void arcneg_fixup(
    const int* __restrict__ label, const float* __restrict__ ws,
    float* __restrict__ out)
{
    const int b = threadIdx.x;
    if (b < BATCH) out[(size_t)b * OUT_F + label[b]] = kSCALE * ws[BATCH + b];
}

extern "C" void kernel_launch(void* const* d_in, const int* in_sizes, int n_in,
                              void* d_out, int out_size, void* d_ws, size_t ws_size,
                              hipStream_t stream)
{
    const float* x     = (const float*)d_in[0];   // [512, 512] f32
    const int*   label = (const int*)  d_in[1];   // [512] i32
    const float* w     = (const float*)d_in[2];   // [100000, 512] f32
    float* out = (float*)d_out;                   // [512, 100000] f32
    float* ws  = (float*)d_ws;                    // [512] xinv + [512] target

    (void)in_sizes; (void)n_in; (void)out_size; (void)ws_size;

    arcneg_prep<<<BATCH, 128, 0, stream>>>(x, label, w, ws);

    dim3 grid(BATCH / BM, (OUT_F + BN - 1) / BN);   // (4, 782), M fastest for L2 reuse
    arcneg_gemm<<<grid, 256, 0, stream>>>(x, w, ws, out);

    arcneg_fixup<<<1, 512, 0, stream>>>(label, ws, out);
}
